// Wave_MLP_2D_24300924961145
// MI455X (gfx1250) — matrix operand order repaired and checked by reading, not compiled, not run on hardware
//
#include <hip/hip_runtime.h>
#include <hip/hip_bf16.h>

typedef __attribute__((ext_vector_type(16))) _Float16 v16h;
typedef __attribute__((ext_vector_type(8)))  _Float16 v8h;
typedef __attribute__((ext_vector_type(8)))  float    v8f;

#define CDIM 64
#define HWSZ 65536
#define NPIX 262144
#define APAD 72      // padded K stride for A/B LDS tiles (bank-conflict avoidance)
#define OPAD 132     // padded pixel stride for output staging

// ---------------------------------------------------------------------------
// WMMA helpers (CDNA5, wave32): D = A(16x32 f16) * B(32x16 f16) + C(16x16 f32)
// ---------------------------------------------------------------------------
__device__ __forceinline__ v8f wmma16(v16h a, v16h b, v8f c) {
  return __builtin_amdgcn_wmma_f32_16x16x32_f16(false, a, false, b, (short)0, c,
                                                false, false);
}

__device__ __forceinline__ v16h ld16h(const _Float16* p) {
  v8h lo = *(const v8h*)p;
  v8h hi = *(const v8h*)(p + 16);
  v16h r;
#pragma unroll
  for (int i = 0; i < 8; ++i) { r[i] = lo[i]; r[i + 8] = hi[i]; }
  return r;
}

__device__ __forceinline__ v8f vzero8() {
  v8f z;
#pragma unroll
  for (int i = 0; i < 8; ++i) z[i] = 0.f;
  return z;
}

// ---------------------------------------------------------------------------
// K0: fold BN params into per-channel scale/shift; zero the pooling buffer.
// fold layout: thS[64] | thT[64] | twS[64] | twT[64]
// ---------------------------------------------------------------------------
__global__ __launch_bounds__(256) void k0_prep(
    const float* __restrict__ th_b, const float* __restrict__ th_g,
    const float* __restrict__ th_be, const float* __restrict__ th_m,
    const float* __restrict__ th_v, const float* __restrict__ tw_b,
    const float* __restrict__ tw_g, const float* __restrict__ tw_be,
    const float* __restrict__ tw_m, const float* __restrict__ tw_v,
    float* __restrict__ fold, float* __restrict__ pool) {
  int t = threadIdx.x;
  if (t < CDIM) {
    float s1 = th_g[t] * rsqrtf(th_v[t] + 1e-5f);
    fold[t] = s1;
    fold[64 + t] = (th_b[t] - th_m[t]) * s1 + th_be[t];
    float s2 = tw_g[t] * rsqrtf(tw_v[t] + 1e-5f);
    fold[128 + t] = s2;
    fold[192 + t] = (tw_b[t] - tw_m[t]) * s2 + tw_be[t];
  }
  pool[t] = 0.f;  // 256 = B*C entries
}

// ---------------------------------------------------------------------------
// K1: fused 5-way 1x1-conv GEMM over x, BN+ReLU+cos/sin, f16 plane outputs.
// Each block = 256 threads (8 waves), 128 consecutive pixels (one batch).
// Each wave: 16-pixel x 64-channel tile; 5 GEMMs x 4 N-tiles x 2 K-steps = 40 WMMA.
// ---------------------------------------------------------------------------
__global__ __launch_bounds__(256) void k1_gemm5(
    const float* __restrict__ x, const float* __restrict__ w_th,
    const float* __restrict__ w_tw, const float* __restrict__ w_fh,
    const float* __restrict__ w_fw, const float* __restrict__ w_fc,
    const float* __restrict__ fold, _Float16* __restrict__ t_ch,
    _Float16* __restrict__ t_sh, _Float16* __restrict__ t_cw,
    _Float16* __restrict__ t_sw, _Float16* __restrict__ t_cc,
    float* __restrict__ pool) {
  __shared__ _Float16 s_w[5][CDIM][APAD];   // weights [gemm][o][c] f16
  __shared__ _Float16 s_a[128][APAD];       // A tile [pixel][c] f16
  __shared__ _Float16 s_out[CDIM][OPAD];    // output staging [o][pixel] f16
  __shared__ float s_pool[CDIM];

  const int tid = threadIdx.x;
  const int wave = tid >> 5;
  const int lane = tid & 31;
  const int P0 = blockIdx.x * 128;
  const int b = P0 >> 16;           // 128 | HW so block stays inside one batch
  const int hw0 = P0 & (HWSZ - 1);

  // ---- stage 5 weight matrices (global [o][c] f32 -> LDS f16) ----
#pragma unroll
  for (int g = 0; g < 5; ++g) {
    const float* W = (g == 0) ? w_th : (g == 1) ? w_tw : (g == 2) ? w_fh
                     : (g == 3) ? w_fw : w_fc;
    for (int r = tid; r < CDIM * CDIM; r += 256)
      s_w[g][r >> 6][r & 63] = (_Float16)W[r];
  }
  // ---- stage x tile: channel-major global -> pixel-major LDS (f16) ----
  const float* xb = x + (size_t)b * (CDIM * HWSZ) + hw0;
#pragma unroll
  for (int i = 0; i < 8; ++i) {
    int cid = tid + 256 * i;         // 2048 float4 chunks (64 ch x 32 chunks)
    int chn = cid >> 5;
    int p4 = (cid & 31) * 4;
    float4 v = *(const float4*)(xb + (size_t)chn * HWSZ + p4);
    s_a[p4 + 0][chn] = (_Float16)v.x;
    s_a[p4 + 1][chn] = (_Float16)v.y;
    s_a[p4 + 2][chn] = (_Float16)v.z;
    s_a[p4 + 3][chn] = (_Float16)v.w;
  }
  if (tid < CDIM) s_pool[tid] = 0.f;
  __syncthreads();

  // ---- build A fragments (ISA 16-bit A 16x32 layout) ----
  const int m = lane & 15;           // A row = pixel within tile; D col = out-ch
  const int hi = lane >> 4;
  const int arow = wave * 16 + m;
  v16h af[2];
#pragma unroll
  for (int kb = 0; kb < 2; ++kb) {
    const _Float16* pa = &s_a[arow][kb * 32 + 8 * hi];
    v8h lo = *(const v8h*)pa;        // K = kb*32 + 8*hi + (0..7)
    v8h hh = *(const v8h*)(pa + 16); // K = kb*32 + 16 + 8*hi + (0..7)
    v16h t;
#pragma unroll
    for (int i = 0; i < 8; ++i) { t[i] = lo[i]; t[i + 8] = hh[i]; }
    af[kb] = t;
  }

  // ---- helpers ----
  auto gemm4 = [&](int g, v8f* acc) {
#pragma unroll
    for (int nt = 0; nt < 4; ++nt) {
      acc[nt] = vzero8();
#pragma unroll
      for (int kb = 0; kb < 2; ++kb) {
        // B layout: lane col = l&15, K = 8*hi+(0..7) and 16+8*hi+(0..7)
        v16h bf = ld16h(&s_w[g][nt * 16 + m][kb * 32 + 8 * hi]);
        acc[nt] = wmma16(af[kb], bf, acc[nt]);
      }
    }
  };
  auto flush = [&](v8f* vals, _Float16* plane) {
#pragma unroll
    for (int nt = 0; nt < 4; ++nt) {
      int o = nt * 16 + m;
      int pp = wave * 16 + 8 * hi;   // D row (pixel) = r + 8*hi
#pragma unroll
      for (int r = 0; r < 8; ++r) s_out[o][pp + r] = (_Float16)vals[nt][r];
    }
    __syncthreads();
#pragma unroll
    for (int i = 0; i < 8; ++i) {    // coalesced 256B/channel plane writes
      int cid = tid + 256 * i;
      int o = cid >> 5;
      int p4 = (cid & 31) * 4;
      *(uint2*)(plane + (size_t)o * NPIX + P0 + p4) = *(const uint2*)&s_out[o][p4];
    }
    __syncthreads();
  };

  const float* thS = fold;
  const float* thT = fold + 64;
  const float* twS = fold + 128;
  const float* twT = fold + 192;

  v8f acc[4], cv[4], sv[4];

  // ---- theta_h = relu(BN(conv(x,th))) ; cos/sin ----
  gemm4(0, acc);
#pragma unroll
  for (int nt = 0; nt < 4; ++nt) {
    float S = thS[nt * 16 + m], T = thT[nt * 16 + m];
#pragma unroll
    for (int r = 0; r < 8; ++r) {
      float t = fmaxf(acc[nt][r] * S + T, 0.f);
      cv[nt][r] = __cosf(t);
      sv[nt][r] = __sinf(t);
    }
  }
  // ---- x_h; emit x_h*cos, x_h*sin ----
  gemm4(2, acc);
#pragma unroll
  for (int nt = 0; nt < 4; ++nt)
#pragma unroll
    for (int r = 0; r < 8; ++r) cv[nt][r] *= acc[nt][r];
  flush(cv, t_ch);
#pragma unroll
  for (int nt = 0; nt < 4; ++nt)
#pragma unroll
    for (int r = 0; r < 8; ++r) sv[nt][r] *= acc[nt][r];
  flush(sv, t_sh);

  // ---- theta_w ----
  gemm4(1, acc);
#pragma unroll
  for (int nt = 0; nt < 4; ++nt) {
    float S = twS[nt * 16 + m], T = twT[nt * 16 + m];
#pragma unroll
    for (int r = 0; r < 8; ++r) {
      float t = fmaxf(acc[nt][r] * S + T, 0.f);
      cv[nt][r] = __cosf(t);
      sv[nt][r] = __sinf(t);
    }
  }
  // ---- x_w; emit x_w*cos, x_w*sin ----
  gemm4(3, acc);
#pragma unroll
  for (int nt = 0; nt < 4; ++nt)
#pragma unroll
    for (int r = 0; r < 8; ++r) cv[nt][r] *= acc[nt][r];
  flush(cv, t_cw);
#pragma unroll
  for (int nt = 0; nt < 4; ++nt)
#pragma unroll
    for (int r = 0; r < 8; ++r) sv[nt][r] *= acc[nt][r];
  flush(sv, t_sw);

  // ---- c = conv(x, fc_c); emit + pool partial sums ----
  gemm4(4, acc);
  flush(acc, t_cc);
#pragma unroll
  for (int nt = 0; nt < 4; ++nt) {
    float s = 0.f;
#pragma unroll
    for (int r = 0; r < 8; ++r) s += acc[nt][r];
    atomicAdd(&s_pool[nt * 16 + m], s);
  }
  __syncthreads();
  if (tid < CDIM) atomicAdd(&pool[b * CDIM + tid], s_pool[tid]);
}

// ---------------------------------------------------------------------------
// K2h: grouped (1,7) conv along W. Group g reads concat channels 2g,2g+1.
// Block = (g, b, 16-row chunk), 256 threads = one row of W.
// ---------------------------------------------------------------------------
__global__ __launch_bounds__(256) void k2_h(
    const _Float16* __restrict__ t_ch, const _Float16* __restrict__ t_sh,
    const float* __restrict__ tfc_h, _Float16* __restrict__ t_h,
    float* __restrict__ pool) {
  __shared__ float s0[262], s1[262];
  __shared__ float s_red[256];
  const int bid = blockIdx.x;
  const int g = bid >> 6;          // 64 = B * (H/16)
  const int rem = bid & 63;
  const int b = rem >> 4;
  const int y0 = (rem & 15) << 4;
  const int t = threadIdx.x;

  const int c2 = 2 * g;
  const _Float16* in0 =
      (c2 < CDIM) ? t_ch + (size_t)c2 * NPIX : t_sh + (size_t)(c2 - CDIM) * NPIX;
  const _Float16* in1 = in0 + NPIX;

  float w0[7], w1[7];
#pragma unroll
  for (int k = 0; k < 7; ++k) {
    w0[k] = tfc_h[g * 14 + k];
    w1[k] = tfc_h[g * 14 + 7 + k];
  }
  float lsum = 0.f;
  for (int y = y0; y < y0 + 16; ++y) {
    size_t rb = (size_t)b * HWSZ + (size_t)y * 256;
    s0[3 + t] = (float)in0[rb + t];
    s1[3 + t] = (float)in1[rb + t];
    if (t < 3) { s0[t] = 0.f; s1[t] = 0.f; s0[259 + t] = 0.f; s1[259 + t] = 0.f; }
    __syncthreads();
    float acc = 0.f;
#pragma unroll
    for (int k = 0; k < 7; ++k) acc += w0[k] * s0[t + k] + w1[k] * s1[t + k];
    t_h[(size_t)g * NPIX + rb + t] = (_Float16)acc;
    lsum += acc;
    __syncthreads();
  }
  s_red[t] = lsum;
  __syncthreads();
  for (int s = 128; s > 0; s >>= 1) {
    if (t < s) s_red[t] += s_red[t + s];
    __syncthreads();
  }
  if (t == 0) atomicAdd(&pool[b * CDIM + g], s_red[0]);
}

// ---------------------------------------------------------------------------
// K2w: grouped (7,1) conv along H with per-thread rolling window.
// Block = (g, b, 64-row chunk), 256 threads = one row of W (column per thread).
// ---------------------------------------------------------------------------
__global__ __launch_bounds__(256) void k2_w(
    const _Float16* __restrict__ t_cw, const _Float16* __restrict__ t_sw,
    const float* __restrict__ tfc_w, _Float16* __restrict__ t_w,
    float* __restrict__ pool) {
  __shared__ float s_red[256];
  const int bid = blockIdx.x;
  const int g = bid >> 4;          // 16 = B * (H/64)
  const int rem = bid & 15;
  const int b = rem >> 2;
  const int y0 = (rem & 3) << 6;
  const int t = threadIdx.x;

  const int c2 = 2 * g;
  const _Float16* in0 =
      (c2 < CDIM) ? t_cw + (size_t)c2 * NPIX : t_sw + (size_t)(c2 - CDIM) * NPIX;
  const _Float16* in1 = in0 + NPIX;

  float w0[7], w1[7];
#pragma unroll
  for (int k = 0; k < 7; ++k) {
    w0[k] = tfc_w[g * 14 + k];
    w1[k] = tfc_w[g * 14 + 7 + k];
  }
  const size_t base = (size_t)b * HWSZ;
  float r0[7], r1[7];
#pragma unroll
  for (int j = 0; j < 7; ++j) {
    int yy = y0 + j - 3;
    bool ok = (yy >= 0) && (yy < 256);
    r0[j] = ok ? (float)in0[base + (size_t)yy * 256 + t] : 0.f;
    r1[j] = ok ? (float)in1[base + (size_t)yy * 256 + t] : 0.f;
  }
  float lsum = 0.f;
  for (int y = y0; y < y0 + 64; ++y) {
    float acc = 0.f;
#pragma unroll
    for (int k = 0; k < 7; ++k) acc += w0[k] * r0[k] + w1[k] * r1[k];
    t_w[(size_t)g * NPIX + base + (size_t)y * 256 + t] = (_Float16)acc;
    lsum += acc;
#pragma unroll
    for (int j = 0; j < 6; ++j) { r0[j] = r0[j + 1]; r1[j] = r1[j + 1]; }
    int yn = y + 4;
    bool ok = yn < 256;
    r0[6] = ok ? (float)in0[base + (size_t)yn * 256 + t] : 0.f;
    r1[6] = ok ? (float)in1[base + (size_t)yn * 256 + t] : 0.f;
  }
  s_red[t] = lsum;
  __syncthreads();
  for (int s = 128; s > 0; s >>= 1) {
    if (t < s) s_red[t] += s_red[t + s];
    __syncthreads();
  }
  if (t == 0) atomicAdd(&pool[b * CDIM + g], s_red[0]);
}

// ---------------------------------------------------------------------------
// K3: reweight MLP + softmax over the 3 branches. Single block.
// aout layout: a[s][b][ch], s in {0,1,2}
// ---------------------------------------------------------------------------
__global__ __launch_bounds__(256) void k3_reweight(
    const float* __restrict__ pool, const float* __restrict__ rw1_w,
    const float* __restrict__ rw1_b, const float* __restrict__ rw2_w,
    const float* __restrict__ rw2_b, float* __restrict__ aout) {
  __shared__ float s_mean[256];   // [B][64]
  __shared__ float s_hid[64];     // [B][16]
  __shared__ float s_o[768];      // [B][192]
  const int t = threadIdx.x;
  s_mean[t] = pool[t] * (1.f / 65536.f);
  __syncthreads();
  if (t < 64) {
    int b = t >> 4, j = t & 15;
    float acc = rw1_b[j];
    for (int c = 0; c < 64; ++c) acc += s_mean[b * 64 + c] * rw1_w[j * 64 + c];
    s_hid[t] = 0.5f * acc * (1.f + erff(acc * 0.70710678118f));  // exact gelu
  }
  __syncthreads();
  for (int idx = t; idx < 768; idx += 256) {
    int b = idx / 192, k = idx % 192;
    float acc = rw2_b[k];
#pragma unroll
    for (int j = 0; j < 16; ++j) acc += s_hid[b * 16 + j] * rw2_w[k * 16 + j];
    s_o[idx] = acc;
  }
  __syncthreads();
  {
    int b = t >> 6, chn = t & 63;
    float e0 = s_o[b * 192 + chn * 3 + 0];
    float e1 = s_o[b * 192 + chn * 3 + 1];
    float e2 = s_o[b * 192 + chn * 3 + 2];
    float mx = fmaxf(e0, fmaxf(e1, e2));
    float x0 = __expf(e0 - mx), x1 = __expf(e1 - mx), x2 = __expf(e2 - mx);
    float inv = 1.f / (x0 + x1 + x2);
    aout[t] = x0 * inv;
    aout[256 + t] = x1 * inv;
    aout[512 + t] = x2 * inv;
  }
}

// ---------------------------------------------------------------------------
// K4: out = conv1x1(h*a0 + w*a1 + c*a2, proj_w) + proj_b  (WMMA GEMM)
// Block = 256 threads, 128 pixels; 8 WMMAs per wave.
// ---------------------------------------------------------------------------
__global__ __launch_bounds__(256) void k4_proj(
    const _Float16* __restrict__ t_h, const _Float16* __restrict__ t_w,
    const _Float16* __restrict__ t_cc, const float* __restrict__ aout,
    const float* __restrict__ proj_w, const float* __restrict__ proj_b,
    float* __restrict__ out) {
  __shared__ _Float16 s_w[CDIM][APAD];
  __shared__ _Float16 s_a[128][APAD];
  __shared__ float s_out[CDIM][OPAD];
  __shared__ float s_asc[3][CDIM];
  __shared__ float s_pb[CDIM];

  const int tid = threadIdx.x;
  const int wave = tid >> 5;
  const int lane = tid & 31;
  const int P0 = blockIdx.x * 128;
  const int b = P0 >> 16;
  const int hw0 = P0 & (HWSZ - 1);

  for (int r = tid; r < CDIM * CDIM; r += 256)
    s_w[r >> 6][r & 63] = (_Float16)proj_w[r];
  if (tid < CDIM) s_pb[tid] = proj_b[tid];
  if (tid < 192) s_asc[tid >> 6][tid & 63] = aout[(tid >> 6) * 256 + b * CDIM + (tid & 63)];
  __syncthreads();

  // stage mixed input (h*a0 + w*a1 + c*a2) into pixel-major f16 A tile
#pragma unroll
  for (int i = 0; i < 8; ++i) {
    int cid = tid + 256 * i;
    int chn = cid >> 5;
    int p4 = (cid & 31) * 4;
    size_t base = (size_t)chn * NPIX + P0 + p4;
    float a0 = s_asc[0][chn], a1 = s_asc[1][chn], a2 = s_asc[2][chn];
#pragma unroll
    for (int j = 0; j < 4; ++j) {
      float v = (float)t_h[base + j] * a0 + (float)t_w[base + j] * a1 +
                (float)t_cc[base + j] * a2;
      s_a[p4 + j][chn] = (_Float16)v;
    }
  }
  __syncthreads();

  const int m = lane & 15;
  const int hi = lane >> 4;
  const int arow = wave * 16 + m;
  v16h af[2];
#pragma unroll
  for (int kb = 0; kb < 2; ++kb) {
    const _Float16* pa = &s_a[arow][kb * 32 + 8 * hi];
    v8h lo = *(const v8h*)pa;
    v8h hh = *(const v8h*)(pa + 16);
    v16h t;
#pragma unroll
    for (int i = 0; i < 8; ++i) { t[i] = lo[i]; t[i + 8] = hh[i]; }
    af[kb] = t;
  }

  v8f acc[4];
#pragma unroll
  for (int nt = 0; nt < 4; ++nt) {
    acc[nt] = vzero8();
#pragma unroll
    for (int kb = 0; kb < 2; ++kb) {
      v16h bf = ld16h(&s_w[nt * 16 + m][kb * 32 + 8 * hi]);
      acc[nt] = wmma16(af[kb], bf, acc[nt]);
    }
  }
  // deposit f32 + bias, transpose through LDS
#pragma unroll
  for (int nt = 0; nt < 4; ++nt) {
    int o = nt * 16 + m;
    int pp = wave * 16 + 8 * hi;
    float bia = s_pb[o];
#pragma unroll
    for (int r = 0; r < 8; ++r) s_out[o][pp + r] = acc[nt][r] + bia;
  }
  __syncthreads();
  float* ob = out + (size_t)b * (CDIM * HWSZ) + hw0;
#pragma unroll
  for (int i = 0; i < 8; ++i) {
    int cid = tid + 256 * i;
    int o = cid >> 5;
    int p4 = (cid & 31) * 4;
    float4 v = *(const float4*)&s_out[o][p4];
    *(float4*)(ob + (size_t)o * HWSZ + p4) = v;
  }
}

// ---------------------------------------------------------------------------
extern "C" void kernel_launch(void* const* d_in, const int* in_sizes, int n_in,
                              void* d_out, int out_size, void* d_ws,
                              size_t ws_size, hipStream_t stream) {
  const float* x      = (const float*)d_in[0];
  const float* fc_h_w = (const float*)d_in[1];
  const float* fc_w_w = (const float*)d_in[2];
  const float* fc_c_w = (const float*)d_in[3];
  const float* th_w   = (const float*)d_in[4];
  const float* th_b   = (const float*)d_in[5];
  const float* th_g   = (const float*)d_in[6];
  const float* th_be  = (const float*)d_in[7];
  const float* th_m   = (const float*)d_in[8];
  const float* th_v   = (const float*)d_in[9];
  const float* tw_w   = (const float*)d_in[10];
  const float* tw_b   = (const float*)d_in[11];
  const float* tw_g   = (const float*)d_in[12];
  const float* tw_be  = (const float*)d_in[13];
  const float* tw_m   = (const float*)d_in[14];
  const float* tw_v   = (const float*)d_in[15];
  const float* tfc_h  = (const float*)d_in[16];
  const float* tfc_w  = (const float*)d_in[17];
  const float* rw1_w  = (const float*)d_in[18];
  const float* rw1_b  = (const float*)d_in[19];
  const float* rw2_w  = (const float*)d_in[20];
  const float* rw2_b  = (const float*)d_in[21];
  const float* proj_w = (const float*)d_in[22];
  const float* proj_b = (const float*)d_in[23];
  float* out = (float*)d_out;

  // workspace layout (f16 planes are [64][NPIX])
  const size_t PLANE = (size_t)CDIM * NPIX * sizeof(_Float16);  // 32 MB
  char* ws = (char*)d_ws;
  _Float16* t_ch = (_Float16*)(ws + 0 * PLANE);
  _Float16* t_sh = (_Float16*)(ws + 1 * PLANE);
  _Float16* t_cw = (_Float16*)(ws + 2 * PLANE);
  _Float16* t_sw = (_Float16*)(ws + 3 * PLANE);
  _Float16* t_cc = (_Float16*)(ws + 4 * PLANE);
  _Float16* t_h  = (_Float16*)(ws + 5 * PLANE);
  _Float16* t_w  = (_Float16*)(ws + 6 * PLANE);
  float* pool = (float*)(ws + 7 * PLANE);             // 256 f32
  float* aout = (float*)(ws + 7 * PLANE + 1024);      // 768 f32
  float* fold = (float*)(ws + 7 * PLANE + 4096 + 1024);  // 256 f32

  k0_prep<<<1, 256, 0, stream>>>(th_b, th_g, th_be, th_m, th_v, tw_b, tw_g,
                                 tw_be, tw_m, tw_v, fold, pool);
  k1_gemm5<<<NPIX / 128, 256, 0, stream>>>(x, th_w, tw_w, fc_h_w, fc_w_w,
                                           fc_c_w, fold, t_ch, t_sh, t_cw,
                                           t_sw, t_cc, pool);
  k2_h<<<CDIM * 4 * 16, 256, 0, stream>>>(t_ch, t_sh, tfc_h, t_h, pool);
  k2_w<<<CDIM * 4 * 4, 256, 0, stream>>>(t_cw, t_sw, tfc_w, t_w, pool);
  k3_reweight<<<1, 256, 0, stream>>>(pool, rw1_w, rw1_b, rw2_w, rw2_b, aout);
  k4_proj<<<NPIX / 128, 256, 0, stream>>>(t_h, t_w, t_cc, aout, proj_w, proj_b,
                                          out);
}
